// DriftDiT_22539988369521
// MI455X (gfx1250) — matrix operand order repaired and checked by reading, not compiled, not run on hardware
//
#include <hip/hip_runtime.h>
#include <hip/hip_bf16.h>

// ---------------------------------------------------------------------------
// Types for CDNA5 WMMA (wave32): v_wmma_f32_16x16x32_bf16
// ---------------------------------------------------------------------------
typedef __bf16 bf8_t   __attribute__((ext_vector_type(8)));
typedef __bf16 bf16v_t __attribute__((ext_vector_type(16)));
typedef float  f8_t    __attribute__((ext_vector_type(8)));

#define NTOK 2048
#define DMODEL 768
#define NHEAD 12
#define HD 64
#define NCOND 16
#define MHID 3072
#define EPSV 1e-6f

__device__ __forceinline__ f8_t wmma_bf16(bf16v_t a, bf16v_t b, f8_t c) {
  return __builtin_amdgcn_wmma_f32_16x16x32_bf16(false, a, false, b, (short)0, c,
                                                 false, false);
}

// A fragment: 16x32 (MxK) bf16. Lane l<16 holds row M=l, K = k..k+7 then k+16..k+23;
// lane l>=16 holds row M=l-16, K = k+8..k+15 then k+24..k+31.  (ISA 7.12.2)
__device__ __forceinline__ bf16v_t load_a_frag(const __bf16* A, int lda, int m,
                                               int k, int lane) {
  int r = lane & 15, hi = lane >> 4;
  const __bf16* p = A + (size_t)(m + r) * lda + k + hi * 8;
  union { bf16v_t v; bf8_t h[2]; } u;
  u.h[0] = *(const bf8_t*)p;
  u.h[1] = *(const bf8_t*)(p + 16);
  return u.v;
}
// Same layout but from LDS.
__device__ __forceinline__ bf16v_t load_a_frag_lds(const __bf16* row0, int ldr,
                                                   int lane) {
  int r = lane & 15, hi = lane >> 4;
  const __bf16* p = row0 + (size_t)r * ldr + hi * 8;
  union { bf16v_t v; bf8_t h[2]; } u;
  u.h[0] = *(const bf8_t*)p;
  u.h[1] = *(const bf8_t*)(p + 16);
  return u.v;
}
// B fragment: 32x16 (KxN) bf16. Lane l holds column N=l&15, K = k+(l>>4)*8 .. +7
// then k+16+(l>>4)*8 .. +7, along K of W[n, :] (B = W^T).  (ISA 7.12.2)
__device__ __forceinline__ bf16v_t load_b_frag(const __bf16* W, int ldw, int n,
                                               int k, int lane) {
  int c = lane & 15, hi = lane >> 4;
  const __bf16* p = W + (size_t)(n + c) * ldw + k + hi * 8;
  union { bf16v_t v; bf8_t h[2]; } u;
  u.h[0] = *(const bf8_t*)p;
  u.h[1] = *(const bf8_t*)(p + 16);
  return u.v;
}

// ---------------------------------------------------------------------------
// Small elementwise / GEMV kernels
// ---------------------------------------------------------------------------
__global__ void cvt_bf16_kernel(const float* __restrict__ src,
                                __bf16* __restrict__ dst, int n) {
  int i = blockIdx.x * 256 + threadIdx.x;
  if (i < n) dst[i] = (__bf16)src[i];
}

__global__ void silu_kernel(const float* __restrict__ c, float* __restrict__ out,
                            int n) {
  int i = blockIdx.x * 256 + threadIdx.x;
  if (i < n) {
    float x = c[i];
    out[i] = x / (1.f + __expf(-x));
  }
}

// mod[o] = sum_k silu_c[k] * ada_w[o*768+k] + ada_b[o]
__global__ void ada_kernel(const float* __restrict__ sc,
                           const float* __restrict__ W,
                           const float* __restrict__ b,
                           float* __restrict__ mod) {
  int o = blockIdx.x * 256 + threadIdx.x;  // 6912 outputs
  const float* wr = W + (size_t)o * DMODEL;
  float acc = 0.f;
  for (int k = 0; k < DMODEL; ++k) acc += sc[k] * wr[k];
  mod[o] = acc + b[o];
}

// h = rms(x,w) [optionally * (1+sc) + sh], written as bf16. One block per row.
__global__ __launch_bounds__(256) void rmsmod_kernel(
    const float* __restrict__ X, const float* __restrict__ w,
    const float* __restrict__ mod, int sh_off, int sc_off,
    __bf16* __restrict__ Hb) {
  int row = blockIdx.x;
  const float* x = X + (size_t)row * DMODEL;
  __shared__ float red[8];
  float vals[3];
  float local = 0.f;
#pragma unroll
  for (int i = 0; i < 3; ++i) {
    int d = threadIdx.x + i * 256;
    vals[i] = x[d];
    local += vals[i] * vals[i];
  }
  for (int o = 16; o; o >>= 1) local += __shfl_xor(local, o, 32);
  if ((threadIdx.x & 31) == 0) red[threadIdx.x >> 5] = local;
  __syncthreads();
  float tot = 0.f;
#pragma unroll
  for (int i = 0; i < 8; ++i) tot += red[i];
  float r = rsqrtf(tot / (float)DMODEL + EPSV);
#pragma unroll
  for (int i = 0; i < 3; ++i) {
    int d = threadIdx.x + i * 256;
    float hv = vals[i] * r * w[d];
    if (sh_off >= 0) hv = hv * (1.f + mod[sc_off + d]) + mod[sh_off + d];
    Hb[(size_t)row * DMODEL + d] = (__bf16)hv;
  }
}

// xout = xin + g[d] * (y + bias[d])
__global__ void residual_kernel(const float* __restrict__ xin,
                                const float* __restrict__ y,
                                const float* __restrict__ bias,
                                const float* __restrict__ mod, int g_off,
                                float* __restrict__ xout) {
  int i = blockIdx.x * 256 + threadIdx.x;  // N*768
  int d = i % DMODEL;
  float b = bias ? bias[d] : 0.f;
  xout[i] = xin[i] + mod[g_off + d] * (y[i] + b);
}

__global__ void swiglu_kernel(const float* __restrict__ u,
                              const float* __restrict__ v,
                              __bf16* __restrict__ s, int n) {
  int i = blockIdx.x * 256 + threadIdx.x;
  if (i < n) {
    float x = u[i];
    s[i] = (__bf16)((x / (1.f + __expf(-x))) * v[i]);
  }
}

// ---------------------------------------------------------------------------
// Generic WMMA GEMM:  C[M,Nout] (f32) = A[M,K] (bf16) @ W[Nout,K]^T (bf16)
// Each wave computes a 16x64 strip. Block = 8 waves = 128 rows.
// ---------------------------------------------------------------------------
__global__ __launch_bounds__(256) void gemm_bf16_kernel(
    const __bf16* __restrict__ A, const __bf16* __restrict__ W,
    float* __restrict__ C, int M, int K, int Nout) {
  int lane = threadIdx.x & 31;
  int wave = threadIdx.x >> 5;
  int m = (blockIdx.y * 8 + wave) * 16;
  int n = blockIdx.x * 64;
  if (m >= M) return;
  f8_t a0 = {}, a1 = {}, a2 = {}, a3 = {};
  for (int k = 0; k < K; k += 32) {
    bf16v_t af = load_a_frag(A, K, m, k, lane);
    bf16v_t b0 = load_b_frag(W, K, n, k, lane);
    bf16v_t b1 = load_b_frag(W, K, n + 16, k, lane);
    bf16v_t b2 = load_b_frag(W, K, n + 32, k, lane);
    bf16v_t b3 = load_b_frag(W, K, n + 48, k, lane);
    a0 = wmma_bf16(af, b0, a0);
    a1 = wmma_bf16(af, b1, a1);
    a2 = wmma_bf16(af, b2, a2);
    a3 = wmma_bf16(af, b3, a3);
  }
  int col = lane & 15, hi = lane >> 4;
#pragma unroll
  for (int j = 0; j < 8; ++j) {
    float* outp = C + (size_t)(m + hi * 8 + j) * Nout + n;
    outp[col] = a0[j];
    outp[16 + col] = a1[j];
    outp[32 + col] = a2[j];
    outp[48 + col] = a3[j];
  }
}

// ---------------------------------------------------------------------------
// qkv prep: per (n,h) wave — RMS(qn/kn) + RoPE on q,k; V transposed to [H][64][N]
// qkv layout [N][3*768]: q at h*64, k at 768+h*64, v at 1536+h*64
// ---------------------------------------------------------------------------
__global__ __launch_bounds__(256) void qkv_prep_kernel(
    const float* __restrict__ qkv, const float* __restrict__ qn_w,
    const float* __restrict__ kn_w, const float* __restrict__ rc,
    const float* __restrict__ rs, __bf16* __restrict__ Qb,
    __bf16* __restrict__ Kb, __bf16* __restrict__ Vt) {
  int lane = threadIdx.x & 31;
  int idx = blockIdx.x * 8 + (threadIdx.x >> 5);  // n*H + h
  int n = idx / NHEAD, h = idx % NHEAD;
  const float* base = qkv + (size_t)n * (3 * DMODEL);
  const float* q = base + h * HD;
  const float* k = base + DMODEL + h * HD;
  const float* v = base + 2 * DMODEL + h * HD;
  float q0 = q[lane], q1 = q[lane + 32];
  float k0 = k[lane], k1 = k[lane + 32];
  float v0 = v[lane], v1 = v[lane + 32];
  float sq = q0 * q0 + q1 * q1;
  float sk = k0 * k0 + k1 * k1;
  for (int o = 16; o; o >>= 1) {
    sq += __shfl_xor(sq, o, 32);
    sk += __shfl_xor(sk, o, 32);
  }
  float rq = rsqrtf(sq / 64.f + EPSV);
  float rk = rsqrtf(sk / 64.f + EPSV);
  q0 = q0 * rq * qn_w[lane]; q1 = q1 * rq * qn_w[lane + 32];
  k0 = k0 * rk * kn_w[lane]; k1 = k1 * rk * kn_w[lane + 32];
  float c0 = rc[n * HD + lane], c1 = rc[n * HD + lane + 32];
  float s0 = rs[n * HD + lane], s1 = rs[n * HD + lane + 32];
  float qo0 = q0 * c0 - q1 * s0, qo1 = q1 * c1 + q0 * s1;
  float ko0 = k0 * c0 - k1 * s0, ko1 = k1 * c1 + k0 * s1;
  size_t off = ((size_t)h * NTOK + n) * HD;
  Qb[off + lane] = (__bf16)qo0; Qb[off + lane + 32] = (__bf16)qo1;
  Kb[off + lane] = (__bf16)ko0; Kb[off + lane + 32] = (__bf16)ko1;
  size_t voff = (size_t)h * HD * NTOK + n;
  Vt[voff + (size_t)lane * NTOK] = (__bf16)v0;
  Vt[voff + (size_t)(lane + 32) * NTOK] = (__bf16)v1;
}

// ---------------------------------------------------------------------------
// Flash self-attention: one wave per 16-row Q tile, stream 32 keys/iter.
// Qb/Kb: [H][N][64] bf16 ; Vt: [H][64][N] bf16 ; Obf: [N][768] bf16
// ---------------------------------------------------------------------------
__global__ __launch_bounds__(256) void flash_sa_kernel(
    const __bf16* __restrict__ Qb, const __bf16* __restrict__ Kb,
    const __bf16* __restrict__ Vt, __bf16* __restrict__ Obf, float scale) {
  __shared__ __attribute__((aligned(16))) float Sls[8][16][32];
  __shared__ __attribute__((aligned(16))) __bf16 Pls[8][16][32];
  __shared__ float Fls[8][16];
  int lane = threadIdx.x & 31;
  int wave = threadIdx.x >> 5;
  int tile = blockIdx.x * 8 + wave;       // 0 .. H*N/16-1
  int h = tile >> 7;                       // /128
  int qm = (tile & 127) * 16;
  const __bf16* Q = Qb + (size_t)h * NTOK * HD;
  const __bf16* Km = Kb + (size_t)h * NTOK * HD;
  const __bf16* V = Vt + (size_t)h * HD * NTOK;
  bf16v_t aq0 = load_a_frag(Q, HD, qm, 0, lane);
  bf16v_t aq1 = load_a_frag(Q, HD, qm, 32, lane);
  f8_t o0 = {}, o1 = {}, o2 = {}, o3 = {};
  float mrun = -1e30f, lrun = 0.f;
  int col = lane & 15, hi = lane >> 4;
  for (int kb = 0; kb < NTOK; kb += 32) {
    f8_t s0 = {}, s1 = {};
    s0 = wmma_bf16(aq0, load_b_frag(Km, HD, kb, 0, lane), s0);
    s0 = wmma_bf16(aq1, load_b_frag(Km, HD, kb, 32, lane), s0);
    s1 = wmma_bf16(aq0, load_b_frag(Km, HD, kb + 16, 0, lane), s1);
    s1 = wmma_bf16(aq1, load_b_frag(Km, HD, kb + 16, 32, lane), s1);
#pragma unroll
    for (int j = 0; j < 8; ++j) {
      Sls[wave][hi * 8 + j][col] = s0[j] * scale;
      Sls[wave][hi * 8 + j][16 + col] = s1[j] * scale;
    }
    asm volatile("s_wait_dscnt 0x0" ::: "memory");
    if (lane < 16) {
      int r = lane;
      float mnew = mrun;
      for (int c2 = 0; c2 < 32; ++c2) mnew = fmaxf(mnew, Sls[wave][r][c2]);
      float fac = __expf(mrun - mnew);
      float lacc = lrun * fac;
      for (int c2 = 0; c2 < 32; ++c2) {
        float p = __expf(Sls[wave][r][c2] - mnew);
        lacc += p;
        Pls[wave][r][c2] = (__bf16)p;
      }
      mrun = mnew;
      lrun = lacc;
      Fls[wave][r] = fac;
    }
    asm volatile("s_wait_dscnt 0x0" ::: "memory");
#pragma unroll
    for (int j = 0; j < 8; ++j) {
      float f = Fls[wave][hi * 8 + j];
      o0[j] *= f; o1[j] *= f; o2[j] *= f; o3[j] *= f;
    }
    bf16v_t ap = load_a_frag_lds(&Pls[wave][0][0], 32, lane);
    o0 = wmma_bf16(ap, load_b_frag(V, NTOK, 0, kb, lane), o0);
    o1 = wmma_bf16(ap, load_b_frag(V, NTOK, 16, kb, lane), o1);
    o2 = wmma_bf16(ap, load_b_frag(V, NTOK, 32, kb, lane), o2);
    o3 = wmma_bf16(ap, load_b_frag(V, NTOK, 48, kb, lane), o3);
  }
  if (lane < 16) Fls[wave][lane] = 1.0f / lrun;
  asm volatile("s_wait_dscnt 0x0" ::: "memory");
#pragma unroll
  for (int j = 0; j < 8; ++j) {
    float f = Fls[wave][hi * 8 + j];
    __bf16* op = Obf + (size_t)(qm + hi * 8 + j) * DMODEL + h * HD;
    op[col] = (__bf16)(o0[j] * f);
    op[16 + col] = (__bf16)(o1[j] * f);
    op[32 + col] = (__bf16)(o2[j] * f);
    op[48 + col] = (__bf16)(o3[j] * f);
  }
}

// ---------------------------------------------------------------------------
// Cross-attention helpers (Nc = 16 keys -> scalar path, negligible FLOPs)
// ---------------------------------------------------------------------------
__global__ __launch_bounds__(256) void ca_kprep_kernel(
    const float* __restrict__ k2, const float* __restrict__ kn_w,
    float* __restrict__ k2n) {  // k2n: [H][16][64]
  int lane = threadIdx.x & 31;
  int idx = blockIdx.x * 8 + (threadIdx.x >> 5);
  if (idx >= NCOND * NHEAD) return;
  int j = idx / NHEAD, h = idx % NHEAD;
  const float* k = k2 + (size_t)j * DMODEL + h * HD;
  float a = k[lane], b = k[lane + 32];
  float s = a * a + b * b;
  for (int o = 16; o; o >>= 1) s += __shfl_xor(s, o, 32);
  float r = rsqrtf(s / 64.f + EPSV);
  float* out = k2n + ((size_t)h * NCOND + j) * HD;
  out[lane] = a * r * kn_w[lane];
  out[lane + 32] = b * r * kn_w[lane + 32];
}

__global__ __launch_bounds__(256) void ca_attn_kernel(
    const float* __restrict__ q2, const float* __restrict__ k2n,
    const float* __restrict__ v2, const float* __restrict__ qn_w,
    const float* __restrict__ rc, const float* __restrict__ rs,
    __bf16* __restrict__ Ob, float scale) {
  __shared__ float qs[8][64];
  __shared__ float ps[8][16];
  int lane = threadIdx.x & 31, wave = threadIdx.x >> 5;
  int idx = blockIdx.x * 8 + wave;
  int n = idx / NHEAD, h = idx % NHEAD;
  const float* q = q2 + (size_t)n * DMODEL + h * HD;
  float a = q[lane], b = q[lane + 32];
  float s = a * a + b * b;
  for (int o = 16; o; o >>= 1) s += __shfl_xor(s, o, 32);
  float r = rsqrtf(s / 64.f + EPSV);
  a = a * r * qn_w[lane];
  b = b * r * qn_w[lane + 32];
  float c0 = rc[n * HD + lane], c1 = rc[n * HD + lane + 32];
  float s0 = rs[n * HD + lane], s1 = rs[n * HD + lane + 32];
  qs[wave][lane] = a * c0 - b * s0;
  qs[wave][lane + 32] = b * c1 + a * s1;
  asm volatile("s_wait_dscnt 0x0" ::: "memory");
  float sc_j = -1e30f;
  if (lane < 16) {
    const float* kk = k2n + ((size_t)h * NCOND + lane) * HD;
    float acc = 0.f;
    for (int d = 0; d < HD; ++d) acc += qs[wave][d] * kk[d];
    sc_j = acc * scale;
  }
  float m = sc_j;
  for (int o = 8; o; o >>= 1) m = fmaxf(m, __shfl_xor(m, o, 32));
  float p = (lane < 16) ? __expf(sc_j - m) : 0.f;
  float lsum = p;
  for (int o = 8; o; o >>= 1) lsum += __shfl_xor(lsum, o, 32);
  if (lane < 16) ps[wave][lane] = p / lsum;
  asm volatile("s_wait_dscnt 0x0" ::: "memory");
  float oa = 0.f, ob = 0.f;
  for (int j = 0; j < NCOND; ++j) {
    float pj = ps[wave][j];
    const float* vv = v2 + (size_t)j * DMODEL + h * HD;
    oa += pj * vv[lane];
    ob += pj * vv[lane + 32];
  }
  size_t off = (size_t)n * DMODEL + h * HD;
  Ob[off + lane] = (__bf16)oa;
  Ob[off + lane + 32] = (__bf16)ob;
}

// ---------------------------------------------------------------------------
// Host orchestration
// ---------------------------------------------------------------------------
extern "C" void kernel_launch(void* const* d_in, const int* in_sizes, int n_in,
                              void* d_out, int out_size, void* d_ws,
                              size_t ws_size, hipStream_t stream) {
  const float* x = (const float*)d_in[0];
  const float* c = (const float*)d_in[1];
  const float* cond = (const float*)d_in[2];
  const float* rope_cos = (const float*)d_in[3];
  const float* rope_sin = (const float*)d_in[4];
  const float* norm1_w = (const float*)d_in[5];
  const float* norm2_w = (const float*)d_in[6];
  const float* norm3_w = (const float*)d_in[7];
  const float* cond_norm_w = (const float*)d_in[8];
  const float* sa_qkv_w = (const float*)d_in[9];
  const float* sa_proj_w = (const float*)d_in[10];
  const float* sa_proj_b = (const float*)d_in[11];
  const float* sa_qn_w = (const float*)d_in[12];
  const float* sa_kn_w = (const float*)d_in[13];
  const float* ca_q_w = (const float*)d_in[14];
  const float* ca_k_w = (const float*)d_in[15];
  const float* ca_v_w = (const float*)d_in[16];
  const float* ca_proj_w = (const float*)d_in[17];
  const float* ca_proj_b = (const float*)d_in[18];
  const float* ca_qn_w = (const float*)d_in[19];
  const float* ca_kn_w = (const float*)d_in[20];
  const float* mlp_w1 = (const float*)d_in[21];
  const float* mlp_w2 = (const float*)d_in[22];
  const float* mlp_w3 = (const float*)d_in[23];
  const float* ada_w = (const float*)d_in[24];
  const float* ada_b = (const float*)d_in[25];

  char* p = (char*)d_ws;
  auto alloc = [&](size_t bytes) -> void* {
    void* r = (void*)p;
    p += (bytes + 255) & ~(size_t)255;
    return r;
  };
  // fp32 scratch
  float* silu_c = (float*)alloc(DMODEL * 4);
  float* mod = (float*)alloc(9 * DMODEL * 4);
  float* qkv = (float*)alloc((size_t)NTOK * 3 * DMODEL * 4);
  float* tmp1 = (float*)alloc((size_t)NTOK * DMODEL * 4);
  float* x1 = (float*)alloc((size_t)NTOK * DMODEL * 4);
  float* x2 = (float*)alloc((size_t)NTOK * DMODEL * 4);
  float* q2 = (float*)alloc((size_t)NTOK * DMODEL * 4);
  float* k2 = (float*)alloc((size_t)NCOND * DMODEL * 4);
  float* v2 = (float*)alloc((size_t)NCOND * DMODEL * 4);
  float* k2n = (float*)alloc((size_t)NHEAD * NCOND * HD * 4);
  float* ubuf = (float*)alloc((size_t)NTOK * MHID * 4);
  float* vbuf = (float*)alloc((size_t)NTOK * MHID * 4);
  // bf16 scratch
  __bf16* wqkv_b = (__bf16*)alloc((size_t)3 * DMODEL * DMODEL * 2);
  __bf16* wsap_b = (__bf16*)alloc((size_t)DMODEL * DMODEL * 2);
  __bf16* wcaq_b = (__bf16*)alloc((size_t)DMODEL * DMODEL * 2);
  __bf16* wcak_b = (__bf16*)alloc((size_t)DMODEL * DMODEL * 2);
  __bf16* wcav_b = (__bf16*)alloc((size_t)DMODEL * DMODEL * 2);
  __bf16* wcap_b = (__bf16*)alloc((size_t)DMODEL * DMODEL * 2);
  __bf16* w1_b = (__bf16*)alloc((size_t)MHID * DMODEL * 2);
  __bf16* w2_b = (__bf16*)alloc((size_t)DMODEL * MHID * 2);
  __bf16* w3_b = (__bf16*)alloc((size_t)MHID * DMODEL * 2);
  __bf16* h_b = (__bf16*)alloc((size_t)NTOK * DMODEL * 2);
  __bf16* q_b = (__bf16*)alloc((size_t)NHEAD * NTOK * HD * 2);
  __bf16* k_b = (__bf16*)alloc((size_t)NHEAD * NTOK * HD * 2);
  __bf16* vt_b = (__bf16*)alloc((size_t)NHEAD * HD * NTOK * 2);
  __bf16* o_b = (__bf16*)alloc((size_t)NTOK * DMODEL * 2);
  __bf16* cn_b = (__bf16*)alloc((size_t)NCOND * DMODEL * 2);
  __bf16* oca_b = (__bf16*)alloc((size_t)NTOK * DMODEL * 2);
  __bf16* s_b = (__bf16*)alloc((size_t)NTOK * MHID * 2);

  const float scale = 0.125f;  // 64^-0.5
  auto cvt = [&](const float* src, __bf16* dst, int n) {
    cvt_bf16_kernel<<<(n + 255) / 256, 256, 0, stream>>>(src, dst, n);
  };
  auto gemm = [&](const __bf16* A, const __bf16* W, float* C, int M, int K,
                  int Nout) {
    dim3 g(Nout / 64, (M + 127) / 128);
    gemm_bf16_kernel<<<g, 256, 0, stream>>>(A, W, C, M, K, Nout);
  };

  // 0) convert weights to bf16
  cvt(sa_qkv_w, wqkv_b, 3 * DMODEL * DMODEL);
  cvt(sa_proj_w, wsap_b, DMODEL * DMODEL);
  cvt(ca_q_w, wcaq_b, DMODEL * DMODEL);
  cvt(ca_k_w, wcak_b, DMODEL * DMODEL);
  cvt(ca_v_w, wcav_b, DMODEL * DMODEL);
  cvt(ca_proj_w, wcap_b, DMODEL * DMODEL);
  cvt(mlp_w1, w1_b, MHID * DMODEL);
  cvt(mlp_w2, w2_b, DMODEL * MHID);
  cvt(mlp_w3, w3_b, MHID * DMODEL);

  // 1) adaLN modulation
  silu_kernel<<<3, 256, 0, stream>>>(c, silu_c, DMODEL);
  ada_kernel<<<9 * DMODEL / 256, 256, 0, stream>>>(silu_c, ada_w, ada_b, mod);

  // 2) self-attention
  rmsmod_kernel<<<NTOK, 256, 0, stream>>>(x, norm1_w, mod, 0, DMODEL, h_b);
  gemm(h_b, wqkv_b, qkv, NTOK, DMODEL, 3 * DMODEL);
  qkv_prep_kernel<<<NTOK * NHEAD / 8, 256, 0, stream>>>(
      qkv, sa_qn_w, sa_kn_w, rope_cos, rope_sin, q_b, k_b, vt_b);
  flash_sa_kernel<<<NHEAD * (NTOK / 16) / 8, 256, 0, stream>>>(q_b, k_b, vt_b,
                                                               o_b, scale);
  gemm(o_b, wsap_b, tmp1, NTOK, DMODEL, DMODEL);
  residual_kernel<<<NTOK * DMODEL / 256, 256, 0, stream>>>(
      x, tmp1, sa_proj_b, mod, 2 * DMODEL, x1);

  // 3) cross-attention
  rmsmod_kernel<<<NTOK, 256, 0, stream>>>(x1, norm2_w, mod, 3 * DMODEL,
                                          4 * DMODEL, h_b);
  rmsmod_kernel<<<NCOND, 256, 0, stream>>>(cond, cond_norm_w, nullptr, -1, -1,
                                           cn_b);
  gemm(h_b, wcaq_b, q2, NTOK, DMODEL, DMODEL);
  gemm(cn_b, wcak_b, k2, NCOND, DMODEL, DMODEL);
  gemm(cn_b, wcav_b, v2, NCOND, DMODEL, DMODEL);
  ca_kprep_kernel<<<(NCOND * NHEAD + 7) / 8, 256, 0, stream>>>(k2, ca_kn_w, k2n);
  ca_attn_kernel<<<NTOK * NHEAD / 8, 256, 0, stream>>>(
      q2, k2n, v2, ca_qn_w, rope_cos, rope_sin, oca_b, scale);
  gemm(oca_b, wcap_b, tmp1, NTOK, DMODEL, DMODEL);
  residual_kernel<<<NTOK * DMODEL / 256, 256, 0, stream>>>(
      x1, tmp1, ca_proj_b, mod, 5 * DMODEL, x2);

  // 4) SwiGLU MLP
  rmsmod_kernel<<<NTOK, 256, 0, stream>>>(x2, norm3_w, mod, 6 * DMODEL,
                                          7 * DMODEL, h_b);
  gemm(h_b, w1_b, ubuf, NTOK, DMODEL, MHID);
  gemm(h_b, w3_b, vbuf, NTOK, DMODEL, MHID);
  swiglu_kernel<<<NTOK * MHID / 256, 256, 0, stream>>>(ubuf, vbuf, s_b,
                                                       NTOK * MHID);
  gemm(s_b, w2_b, tmp1, NTOK, MHID, DMODEL);
  residual_kernel<<<NTOK * DMODEL / 256, 256, 0, stream>>>(
      x2, tmp1, nullptr, mod, 8 * DMODEL, (float*)d_out);

  (void)in_sizes; (void)n_in; (void)out_size; (void)ws_size;
}
